// TopDownGAT_85031762526449
// MI455X (gfx1250) — matrix operand order repaired and checked by reading, not compiled, not run on hardware
//
#include <hip/hip_runtime.h>
#include <stdint.h>

// Problem constants (match reference).
#define N_NODES 50000
#define E_EDGES 800000
#define ETOT (E_EDGES + N_NODES)   // 850000 (self-loops appended)
#define IN_DIM 256
#define HID 64
#define HEADS 4
#define F (HEADS * HID)            // 256
#define SLOPE 0.2f

// ---- CDNA5 WMMA types (wave32) ----
typedef __attribute__((ext_vector_type(16))) __bf16 v16bf;
typedef __attribute__((ext_vector_type(8)))  float  v8f;

union Frag { v16bf v; uint4 u[2]; };

// fp32 -> bf16 round-to-nearest-even (bit-level; avoids relying on __bf16 arith)
__device__ __forceinline__ unsigned short f2bf(float f) {
  unsigned u = __float_as_uint(f);
  u += 0x7fffu + ((u >> 16) & 1u);
  return (unsigned short)(u >> 16);
}

// Monotonic int encoding of IEEE float so atomicMax(int) == float max.
__device__ __forceinline__ int encf(float f) {
  int i = __float_as_int(f);
  return i >= 0 ? i : (i ^ 0x7fffffff);
}
__device__ __forceinline__ float decf(int e) {
  int i = e >= 0 ? e : (e ^ 0x7fffffff);
  return __int_as_float(i);
}

// ---------------- elementwise helpers ----------------
__global__ void k_f32_to_bf16(const float* __restrict__ in,
                              unsigned short* __restrict__ out, int n) {
  int i = blockIdx.x * blockDim.x + threadIdx.x;
  if (i < n) out[i] = f2bf(in[i]);
}

__global__ void k_fill_f32(float* __restrict__ p, float v, int n) {
  int i = blockIdx.x * blockDim.x + threadIdx.x;
  if (i < n) p[i] = v;
}

__global__ void k_fill_i32(int* __restrict__ p, int v, int n) {
  int i = blockIdx.x * blockDim.x + threadIdx.x;
  if (i < n) p[i] = v;
}

// ---------------- WMMA GEMM: C[M,Nc] = A[M,K](bf16) @ W[Nc,K](bf16)^T ----------------
// Register blocking: one wave computes a 16x64 C strip (4 accumulators), reusing
// one A fragment across 4 B fragments -> 4x fewer A loads, 4 WMMAs per K-step.
// blockDim.x = 32 * (Nc/64); wave w covers cols [w*64, w*64+64). grid.x = M/16.
// K % 32 == 0, all tile math exact -> EXEC stays all-ones (WMMA requirement).
__global__ void __launch_bounds__(128)
k_wmma_gemm_bf16(const unsigned short* __restrict__ A,
                 const unsigned short* __restrict__ Bw,
                 float* __restrict__ C, int K, int Nc) {
  const int lane  = threadIdx.x & 31;
  const int wave  = threadIdx.x >> 5;
  const int tileM = blockIdx.x * 16;
  const int tileN = wave * 64;
  const int half  = lane >> 4;     // 0: lanes 0-15, 1: lanes 16-31
  const int l     = lane & 15;

  // A fragment (16x32 bf16, MxK): lane half 0 holds K 0..15, half 1 holds 16..31
  const unsigned short* arow = A + (size_t)(tileM + l) * K;
  // B fragment (32x16 bf16, KxN): N = col + l; half 0 holds K 0..15, half 1 holds 16..31.
  // B[k][n] = W[n][k], W row-major -> each lane reads 16 contiguous bf16 of W's row.
  const unsigned short* brow = Bw + (size_t)(tileN + l) * K + half * 16;

  v8f acc[4];
#pragma unroll
  for (int j = 0; j < 4; ++j) acc[j] = (v8f){0.f, 0.f, 0.f, 0.f, 0.f, 0.f, 0.f, 0.f};

  for (int kk = 0; kk < K; kk += 32) {
    Frag a;
    a.u[0] = *reinterpret_cast<const uint4*>(arow + kk + half * 16);
    a.u[1] = *reinterpret_cast<const uint4*>(arow + kk + half * 16 + 8);
#pragma unroll
    for (int j = 0; j < 4; ++j) {
      Frag b;
      const unsigned short* bp = brow + (size_t)j * 16 * K + kk;
      b.u[0] = *reinterpret_cast<const uint4*>(bp);
      b.u[1] = *reinterpret_cast<const uint4*>(bp + 8);
      acc[j] = __builtin_amdgcn_wmma_f32_16x16x32_bf16(false, a.v, false, b.v,
                                                       (short)0, acc[j], false, false);
    }
  }
  // C/D layout: VGPR r -> (M = tileM + half*8 + r, N = col + l)
#pragma unroll
  for (int j = 0; j < 4; ++j) {
    float* crow = C + (size_t)(tileM + half * 8) * Nc + tileN + j * 16 + l;
#pragma unroll
    for (int r = 0; r < 8; ++r) crow[(size_t)r * Nc] = acc[j][r];
  }
}

// ---------------- attention coefficients: as/ad[n,h] = <h[n,h,:], a_src/dst[h,:]> ----------------
__global__ void k_attn_coeff(const float* __restrict__ h,
                             const float* __restrict__ a_src,
                             const float* __restrict__ a_dst,
                             float* __restrict__ as_, float* __restrict__ ad_, int n_nh) {
  int idx = blockIdx.x * blockDim.x + threadIdx.x;
  if (idx >= n_nh) return;
  int node = idx >> 2, hd = idx & 3;
  const float* hr = h + (size_t)node * F + hd * HID;
  const float* sr = a_src + hd * HID;
  const float* dr = a_dst + hd * HID;
  float s = 0.f, d = 0.f;
#pragma unroll 8
  for (int c = 0; c < HID; ++c) { float v = hr[c]; s += v * sr[c]; d += v * dr[c]; }
  as_[idx] = s; ad_[idx] = d;
}

// ---------------- edge pass 1: segment max of leaky-relu logits ----------------
__global__ void k_edge_max(const int* __restrict__ ei,
                           const float* __restrict__ as_, const float* __restrict__ ad_,
                           int* __restrict__ mEnc) {
  int idx = blockIdx.x * blockDim.x + threadIdx.x;
  if (idx >= ETOT * HEADS) return;
  int e = idx >> 2, hd = idx & 3;
  int src = (e < E_EDGES) ? ei[e] : (e - E_EDGES);
  int dst = (e < E_EDGES) ? ei[E_EDGES + e] : (e - E_EDGES);
  float v = as_[src * HEADS + hd] + ad_[dst * HEADS + hd];
  v = v > 0.f ? v : SLOPE * v;
  atomicMax(&mEnc[dst * HEADS + hd], encf(v));
}

// ---------------- edge pass 2: exp(logit - max), segment denom ----------------
__global__ void k_edge_exp(const int* __restrict__ ei,
                           const float* __restrict__ as_, const float* __restrict__ ad_,
                           const int* __restrict__ mEnc, float* __restrict__ den,
                           float* __restrict__ ebuf) {
  int idx = blockIdx.x * blockDim.x + threadIdx.x;
  if (idx >= ETOT * HEADS) return;
  int e = idx >> 2, hd = idx & 3;
  int src = (e < E_EDGES) ? ei[e] : (e - E_EDGES);
  int dst = (e < E_EDGES) ? ei[E_EDGES + e] : (e - E_EDGES);
  float v = as_[src * HEADS + hd] + ad_[dst * HEADS + hd];
  v = v > 0.f ? v : SLOPE * v;
  float m  = decf(mEnc[dst * HEADS + hd]);
  float ev = __expf(v - m);
  ebuf[idx] = ev;
  atomicAdd(&den[dst * HEADS + hd], ev);
}

// ---------------- edge pass 3: weighted scatter, one block (256 thr) per edge ----------------
__global__ void __launch_bounds__(256)
k_edge_agg(const int* __restrict__ ei, const float* __restrict__ ebuf,
           const float* __restrict__ den, const float* __restrict__ h,
           float* __restrict__ agg) {
  int e = blockIdx.x;
  int t = threadIdx.x;           // feature index 0..255
  int hd = t >> 6;               // head
  int src = (e < E_EDGES) ? ei[e] : (e - E_EDGES);
  int dst = (e < E_EDGES) ? ei[E_EDGES + e] : (e - E_EDGES);
  float w = ebuf[e * HEADS + hd] / den[dst * HEADS + hd];
  atomicAdd(&agg[(size_t)dst * F + t], w * h[(size_t)src * F + t]);
}

// ---------------- layer-1 epilogue: relu(agg + b1) -> bf16 activations ----------------
__global__ void k_finish1(const float* __restrict__ agg, const float* __restrict__ b1,
                          unsigned short* __restrict__ actb, int n) {
  int i = blockIdx.x * blockDim.x + threadIdx.x;
  if (i >= n) return;
  float v = agg[i] + b1[i & (F - 1)];
  actb[i] = f2bf(v > 0.f ? v : 0.f);
}

// ---------------- final fuse: head-mean + b2 + root-proj + br ----------------
__global__ void k_finish2(const float* __restrict__ agg, const float* __restrict__ b2,
                          const float* __restrict__ rootp, const float* __restrict__ br,
                          float* __restrict__ out, int n) {
  int i = blockIdx.x * blockDim.x + threadIdx.x;
  if (i >= n) return;
  int node = i >> 6, c = i & (HID - 1);
  const float* a = agg + (size_t)node * F + c;
  float m = 0.25f * (a[0] + a[HID] + a[2 * HID] + a[3 * HID]);
  out[i] = m + b2[c] + rootp[i] + br[c];
}

// =====================================================================
extern "C" void kernel_launch(void* const* d_in, const int* in_sizes, int n_in,
                              void* d_out, int out_size, void* d_ws, size_t ws_size,
                              hipStream_t stream) {
  const float* x     = (const float*)d_in[0];
  const float* root  = (const float*)d_in[1];
  const int*   ei    = (const int*)d_in[2];
  const float* W1    = (const float*)d_in[3];
  const float* as1   = (const float*)d_in[4];
  const float* ad1   = (const float*)d_in[5];
  const float* b1    = (const float*)d_in[6];
  const float* W2    = (const float*)d_in[7];
  const float* as2   = (const float*)d_in[8];
  const float* ad2   = (const float*)d_in[9];
  const float* b2    = (const float*)d_in[10];
  const float* Wr    = (const float*)d_in[11];
  const float* br    = (const float*)d_in[12];
  float* out = (float*)d_out;

  // Workspace carve-up (~158 MB total), 256-byte aligned slices.
  char* w = (char*)d_ws;
  size_t off = 0;
  auto carve = [&](size_t bytes) -> void* {
    off = (off + 255) & ~(size_t)255;
    void* p = w + off; off += bytes; return p;
  };
  float*          hbuf  = (float*)carve((size_t)N_NODES * F * 4);       // transformed features
  float*          agg   = (float*)carve((size_t)N_NODES * F * 4);       // aggregation target
  float*          asb   = (float*)carve((size_t)N_NODES * HEADS * 4);
  float*          adb   = (float*)carve((size_t)N_NODES * HEADS * 4);
  int*            mEnc  = (int*)  carve((size_t)N_NODES * HEADS * 4);
  float*          den   = (float*)carve((size_t)N_NODES * HEADS * 4);
  float*          ebuf  = (float*)carve((size_t)ETOT * HEADS * 4);
  float*          rootp = (float*)carve((size_t)N_NODES * HID * 4);
  unsigned short* actb  = (unsigned short*)carve((size_t)N_NODES * F * 2);
  unsigned short* wb    = (unsigned short*)carve((size_t)F * F * 2);

  const int T = 256;
  auto blks = [&](int n) { return (n + T - 1) / T; };
  const int nNF  = N_NODES * F;        // 12.8M
  const int nNH  = N_NODES * HEADS;    // 200k
  const int nEH  = ETOT * HEADS;       // 3.4M
  const int nOut = N_NODES * HID;      // 3.2M

  dim3 gemmGrid(N_NODES / 16);         // one block per 16-row stripe
  dim3 gemmBlkF(32 * (F / 64));        // 128 thr: 4 waves tile 256 cols
  dim3 gemmBlkR(32 * (HID / 64));      // 32 thr: 1 wave tiles 64 cols

  // ---- Layer 1 ----
  k_f32_to_bf16<<<blks(F * F), T, 0, stream>>>(W1, wb, F * F);
  k_f32_to_bf16<<<blks(nNF), T, 0, stream>>>(x, actb, nNF);
  k_wmma_gemm_bf16<<<gemmGrid, gemmBlkF, 0, stream>>>(actb, wb, hbuf, IN_DIM, F);
  k_attn_coeff<<<blks(nNH), T, 0, stream>>>(hbuf, as1, ad1, asb, adb, nNH);
  k_fill_i32<<<blks(nNH), T, 0, stream>>>(mEnc, (int)0x80000000, nNH);
  k_fill_f32<<<blks(nNH), T, 0, stream>>>(den, 0.f, nNH);
  k_fill_f32<<<blks(nNF), T, 0, stream>>>(agg, 0.f, nNF);
  k_edge_max<<<blks(nEH), T, 0, stream>>>(ei, asb, adb, mEnc);
  k_edge_exp<<<blks(nEH), T, 0, stream>>>(ei, asb, adb, mEnc, den, ebuf);
  k_edge_agg<<<ETOT, 256, 0, stream>>>(ei, ebuf, den, hbuf, agg);
  k_finish1<<<blks(nNF), T, 0, stream>>>(agg, b1, actb, nNF);  // relu + bf16 for layer 2

  // ---- Layer 2 ----
  k_f32_to_bf16<<<blks(F * F), T, 0, stream>>>(W2, wb, F * F);
  k_wmma_gemm_bf16<<<gemmGrid, gemmBlkF, 0, stream>>>(actb, wb, hbuf, F, F);
  k_attn_coeff<<<blks(nNH), T, 0, stream>>>(hbuf, as2, ad2, asb, adb, nNH);
  k_fill_i32<<<blks(nNH), T, 0, stream>>>(mEnc, (int)0x80000000, nNH);
  k_fill_f32<<<blks(nNH), T, 0, stream>>>(den, 0.f, nNH);
  k_fill_f32<<<blks(nNF), T, 0, stream>>>(agg, 0.f, nNF);
  k_edge_max<<<blks(nEH), T, 0, stream>>>(ei, asb, adb, mEnc);
  k_edge_exp<<<blks(nEH), T, 0, stream>>>(ei, asb, adb, mEnc, den, ebuf);
  k_edge_agg<<<ETOT, 256, 0, stream>>>(ei, ebuf, den, hbuf, agg);

  // ---- Root projection + final fuse ----
  k_f32_to_bf16<<<blks(HID * IN_DIM), T, 0, stream>>>(Wr, wb, HID * IN_DIM);
  k_f32_to_bf16<<<blks(nNF), T, 0, stream>>>(root, actb, nNF);
  k_wmma_gemm_bf16<<<gemmGrid, gemmBlkR, 0, stream>>>(actb, wb, rootp, IN_DIM, HID);
  k_finish2<<<blks(nOut), T, 0, stream>>>(agg, b2, rootp, br, out, nOut);
}
